// MultiheadedAttention_71846212928240
// MI455X (gfx1250) — matrix operand order repaired and checked by reading, not compiled, not run on hardware
//
#include <hip/hip_runtime.h>
#include <math.h>

#define B_  2
#define N_  2048
#define DM  1024
#define H_  16
#define DH  64

typedef __attribute__((ext_vector_type(16))) _Float16 v16h;
typedef __attribute__((ext_vector_type(8)))  _Float16 v8h;
typedef __attribute__((ext_vector_type(8)))  float    v8f;
typedef __attribute__((ext_vector_type(4)))  unsigned int u32x4;
typedef __attribute__((ext_vector_type(8)))  int          i32x8;
typedef __attribute__((ext_vector_type(4)))  int          i32x4;

#if __has_builtin(__builtin_amdgcn_tensor_load_to_lds)
#define USE_TDM 1
#else
#define USE_TDM 0
#endif

#if __has_builtin(__builtin_amdgcn_s_wait_tensorcnt)
#define WAIT_TENSOR(n) __builtin_amdgcn_s_wait_tensorcnt((short)(n))
#else
#define WAIT_TENSOR(n) asm volatile("s_wait_tensorcnt %0" ::"i"(n) : "memory")
#endif

// ---------------------------------------------------------------------------
// WMMA wrapper: D = A(16x32 f16) * B(32x16 f16) + C(16x16 f32)
// ---------------------------------------------------------------------------
__device__ __forceinline__ v8f wmma_f16(v16h a, v16h b, v8f c) {
  return __builtin_amdgcn_wmma_f32_16x16x32_f16(
      false, a, false, b, (short)0, c, false, false);
}

// A-fragment (16x32, MxK), row-major source.
// lanes 0-15 : K = {0..15};  lanes 16-31: K = {16..31}
__device__ __forceinline__ v16h ld_frag_a(const _Float16* rowp, unsigned lane) {
  const unsigned off = (lane >> 4) * 16u;
  v8h lo = *(const v8h*)(rowp + off);
  v8h hi = *(const v8h*)(rowp + off + 8);
  v16h r;
#pragma unroll
  for (int i = 0; i < 8; ++i) { r[i] = lo[i]; r[i + 8] = hi[i]; }
  return r;
}

// B-fragment (32x16, KxN): lane holds col N=(lane&15), 16 consecutive K halfs,
// K-offset 16*(lane>>4).  p includes row + K offsets already.
__device__ __forceinline__ v16h ld_frag_b(const _Float16* p) {
  return *(const v16h*)p;  // 32 contiguous bytes
}

#if USE_TDM
__device__ __forceinline__ int rfl(unsigned x) {
  return __builtin_amdgcn_readfirstlane((int)x);
}
// Generic pointers to LDS carry the LDS byte offset in addr[31:0] (ISA §10.2).
__device__ __forceinline__ unsigned lds_off(const void* p) {
  return (unsigned)(unsigned long long)p;
}
// 2D TDM load: tile_d1 rows of tile_d0 8-byte units, row stride `stride`
// (8-byte units), packed contiguously into LDS at lds byte address.
__device__ __forceinline__ void tdm_load(unsigned lds, const void* gptr,
                                         unsigned tile_d0, unsigned tile_d1,
                                         unsigned stride) {
  const unsigned long long ga = (unsigned long long)gptr;
  u32x4 g0;
  g0[0] = (unsigned)rfl(1u);                         // count=1, user mode
  g0[1] = (unsigned)rfl(lds);                        // lds_addr [63:32]
  g0[2] = (unsigned)rfl((unsigned)(ga & 0xffffffffu));     // global_addr lo
  g0[3] = (unsigned)rfl((unsigned)((ga >> 32) & 0x1ffffffu) | (2u << 30)); // hi + type=2
  const unsigned td0 = 1u << 20, td1 = 1u << 20;     // tensor dims: no OOB clip
  i32x8 g1;
  g1[0] = rfl(3u << 16);                             // wg_mask=0, data_size=8B
  g1[1] = rfl((td0 & 0xffffu) << 16);                // tensor_dim0[15:0] @ 63:48
  g1[2] = rfl(((td0 >> 16) & 0xffffu) | ((td1 & 0xffffu) << 16));
  g1[3] = rfl(((td1 >> 16) & 0xffffu) | ((tile_d0 & 0xffffu) << 16));
  g1[4] = rfl(tile_d1 & 0xffffu);                    // tile_dim1; tile_dim2=0
  g1[5] = rfl(stride);                               // tensor_dim0_stride lo
  g1[6] = rfl(0u);                                   // stride hi, dim1_stride lo
  g1[7] = rfl(0u);
  i32x4 z4 = {};
#if __has_include(<hip/amd_detail/amd_gfx1250_TDM.h>)
  i32x8 z8 = {};
  __builtin_amdgcn_tensor_load_to_lds(g0, g1, z4, z4, z8, 0);
#else
  __builtin_amdgcn_tensor_load_to_lds(g0, g1, z4, z4, 0);
#endif
}
#endif  // USE_TDM

// ---------------------------------------------------------------------------
// Stage 1a: fp32 -> f16 elementwise
// ---------------------------------------------------------------------------
__global__ void cvt_f32_f16_kernel(const float* __restrict__ x,
                                   _Float16* __restrict__ y, int n) {
  for (int i = blockIdx.x * blockDim.x + threadIdx.x; i < n;
       i += gridDim.x * blockDim.x)
    y[i] = (_Float16)x[i];
}

// Stage 1b: W[h][d][e] fp32 -> Wt[h][e][d] f16 (B-operand layout)
__global__ void wtr_kernel(const float* __restrict__ w, _Float16* __restrict__ wt) {
  const int n = H_ * DM * DH;
  for (int i = blockIdx.x * blockDim.x + threadIdx.x; i < n;
       i += gridDim.x * blockDim.x) {
    const int h   = i / (DM * DH);
    const int rem = i - h * DM * DH;
    const int e   = rem / DM;
    const int d   = rem - e * DM;
    wt[i] = (_Float16)w[(h * DM + d) * DH + e];
  }
}

// ---------------------------------------------------------------------------
// Stage 2: projection GEMM.  One wave -> 16 tokens x 64 outputs for one head.
// ---------------------------------------------------------------------------
__global__ __launch_bounds__(128) void proj_kernel(
    const _Float16* __restrict__ Xh, const _Float16* __restrict__ Wt,
    const float* __restrict__ bias, _Float16* __restrict__ out,
    int v_transposed, float scale) {
  const unsigned lane = threadIdx.x & 31u;
  const unsigned warp = threadIdx.x >> 5;
  const unsigned wid  = blockIdx.x * 4u + warp;   // 0..4095
  const unsigned b  = wid >> 11;
  const unsigned h  = (wid >> 7) & 15u;
  const unsigned t  = wid & 127u;

  const _Float16* xrow = Xh + (size_t)(b * N_ + t * 16u + (lane & 15u)) * DM;
  const _Float16* wrow =
      Wt + ((size_t)(h * DH) + (lane & 15u)) * DM + (lane >> 4) * 16u;

  v8f acc[4] = {};
#pragma unroll 4
  for (int d = 0; d < DM; d += 32) {
    v16h a = ld_frag_a(xrow + d, lane);
#pragma unroll
    for (int et = 0; et < 4; ++et) {
      v16h bb = ld_frag_b(wrow + (size_t)et * 16u * DM + d);
      acc[et] = wmma_f16(a, bb, acc[et]);
    }
  }

  const unsigned col    = lane & 15u;
  const unsigned rowoff = 8u * (lane >> 4);
#pragma unroll
  for (int et = 0; et < 4; ++et) {
    const float bv = bias[h * DH + et * 16 + col];
    if (!v_transposed) {
      _Float16* op = out +
          ((size_t)(b * H_ + h) * N_ + t * 16u + rowoff) * DH + et * 16 + col;
#pragma unroll
      for (int r = 0; r < 8; ++r)
        op[(size_t)r * DH] = (_Float16)((acc[et][r] + bv) * scale);
    } else {
      v8h pk;
#pragma unroll
      for (int r = 0; r < 8; ++r) pk[r] = (_Float16)((acc[et][r] + bv) * scale);
      _Float16* op = out +
          ((size_t)(b * H_ + h) * DH + et * 16 + col) * N_ + t * 16u + rowoff;
      *(v8h*)op = pk;
    }
  }
}

// ---------------------------------------------------------------------------
// Stage 3: causal flash attention.  One wave per (b,h,16-query tile).
// K/V tiles staged into LDS by the Tensor Data Mover, double-buffered,
// synchronized with s_wait_tensorcnt.  4 WMMA for S, online softmax,
// P transposed through LDS, 4 WMMA for O += P*V.
// ---------------------------------------------------------------------------
#define KT_HALFS (32 * 64)   // 32 kv rows x 64 dims  (contiguous 4KB)
#define VT_HALFS (64 * 32)   // 64 e rows x 32 kv     (strided source)

__global__ __launch_bounds__(64) void attn_kernel(
    const _Float16* __restrict__ Qh, const _Float16* __restrict__ Kh,
    const _Float16* __restrict__ Vt, float* __restrict__ out) {
  __shared__ __align__(32) _Float16 ldsK[2][2][KT_HALFS];
  __shared__ __align__(32) _Float16 ldsV[2][2][VT_HALFS];
  __shared__ __align__(32) _Float16 ldsP[2][16 * 32];

  const unsigned lane = threadIdx.x & 31u;
  const unsigned warp = threadIdx.x >> 5;          // 0..1
  const unsigned wid  = blockIdx.x * 2u + warp;    // 0..4095
  const unsigned b  = wid >> 11;
  const unsigned h  = (wid >> 7) & 15u;
  const unsigned qt = wid & 127u;
  const int qbase = (int)qt * 16;

  const size_t bh = (size_t)(b * H_ + h);
  const _Float16* qrow = Qh + (bh * N_ + (size_t)qbase + (lane & 15u)) * DH;
  const v16h aq0 = ld_frag_a(qrow, lane);
  const v16h aq1 = ld_frag_a(qrow + 32, lane);

  const _Float16* kbase = Kh + bh * (size_t)N_ * DH;
  const _Float16* vbase = Vt + bh * (size_t)DH * N_;

  float m[8], l[8];
  v8f o[4] = {};
#pragma unroll
  for (int r = 0; r < 8; ++r) { m[r] = -INFINITY; l[r] = 0.0f; }

  const unsigned col  = lane & 15u;
  const unsigned half = lane >> 4;
  const int qrow0 = qbase + (int)(half * 8u);
  _Float16*       pst = &ldsP[warp][col];
  const _Float16* pld = &ldsP[warp][col * 32u];

#if USE_TDM
  // K tile: one row of 512 x 8B (contiguous 4KB).  V tile: 64 rows x 8 x 8B,
  // source row stride = N_*2 bytes = 512 x 8B.
  tdm_load(lds_off(&ldsK[warp][0][0]), kbase, 512u, 1u, 512u);
  tdm_load(lds_off(&ldsV[warp][0][0]), vbase, 8u, 64u, 512u);
#endif

  for (int jb = 0; jb <= qbase; jb += 32) {
#if USE_TDM
    const int cur = (jb >> 5) & 1;
    if (jb + 32 <= qbase) {
      tdm_load(lds_off(&ldsK[warp][cur ^ 1][0]),
               kbase + (size_t)(jb + 32) * DH, 512u, 1u, 512u);
      tdm_load(lds_off(&ldsV[warp][cur ^ 1][0]),
               vbase + (jb + 32), 8u, 64u, 512u);
      WAIT_TENSOR(2);          // current tile's 2 TDM ops completed (in-order)
    } else {
      WAIT_TENSOR(0);
    }
    asm volatile("" ::: "memory");
#else
    const int cur = 0;
    {  // synchronous per-lane staging fallback
      const _Float16* gk = kbase + (size_t)jb * DH;
      for (int c = (int)lane; c < KT_HALFS / 8; c += 32)
        *(v8h*)(&ldsK[warp][0][c * 8]) = *(const v8h*)(gk + c * 8);
      const _Float16* gv = vbase + jb;
      for (int c = (int)lane; c < VT_HALFS / 8; c += 32) {
        const int r = c >> 2, off = (c & 3) * 8;
        *(v8h*)(&ldsV[warp][0][r * 32 + off]) =
            *(const v8h*)(gv + (size_t)r * N_ + off);
      }
      asm volatile("s_wait_dscnt 0" ::: "memory");
    }
#endif

    // ---- S = Q * K^T from LDS K tile ----
    const _Float16* kt = &ldsK[warp][cur][0];
    const _Float16* k0 = kt + (size_t)col * DH + half * 16u;   // kv row = col
    const _Float16* k1 = k0 + 16 * DH;                         // kv row = col+16
    v8f s0 = {}, s1 = {};
    s0 = wmma_f16(aq0, ld_frag_b(k0),      s0);
    s0 = wmma_f16(aq1, ld_frag_b(k0 + 32), s0);
    s1 = wmma_f16(aq0, ld_frag_b(k1),      s1);
    s1 = wmma_f16(aq1, ld_frag_b(k1 + 32), s1);

    // ---- causal mask + online softmax ----
    const int kv0 = jb + (int)col, kv1 = kv0 + 16;
#pragma unroll
    for (int r = 0; r < 8; ++r) {
      const int q = qrow0 + r;
      float a0 = (kv0 <= q) ? s0[r] : -1.0e30f;
      float a1 = (kv1 <= q) ? s1[r] : -1.0e30f;
      float tm = fmaxf(a0, a1);
#pragma unroll
      for (int msk = 1; msk < 16; msk <<= 1)
        tm = fmaxf(tm, __shfl_xor(tm, msk, 32));
      const float nm  = fmaxf(m[r], tm);
      const float fac = __expf(m[r] - nm);
      m[r] = nm;
      const float p0 = __expf(a0 - nm);
      const float p1 = __expf(a1 - nm);
      float ts = p0 + p1;
#pragma unroll
      for (int msk = 1; msk < 16; msk <<= 1) ts += __shfl_xor(ts, msk, 32);
      l[r] = l[r] * fac + ts;
#pragma unroll
      for (int n = 0; n < 4; ++n) o[n][r] *= fac;
      const int prow = (r + (int)(half * 8u)) * 32;
      pst[prow]      = (_Float16)p0;
      pst[prow + 16] = (_Float16)p1;
    }

    asm volatile("s_wait_dscnt 0" ::: "memory");
    const v16h pa = ld_frag_a(pld, lane);          // P as A-fragment (16x32)

    // ---- O += P * V from LDS V tile ----
#pragma unroll
    for (int n = 0; n < 4; ++n) {
      const _Float16* vp = &ldsV[warp][cur][(n * 16 + col) * 32 + half * 16u];
      o[n] = wmma_f16(pa, ld_frag_b(vp), o[n]);
    }
  }

  float inv[8];
#pragma unroll
  for (int r = 0; r < 8; ++r) inv[r] = 1.0f / l[r];

  // out[b][n][h*64 + e], fp32
#pragma unroll
  for (int et = 0; et < 4; ++et) {
    float* op = out + ((size_t)b * N_ + (size_t)qbase + half * 8u) * DM +
                h * DH + et * 16 + col;
#pragma unroll
    for (int r = 0; r < 8; ++r) op[(size_t)r * DM] = o[et][r] * inv[r];
  }
}

// ---------------------------------------------------------------------------
// Host launcher
// ---------------------------------------------------------------------------
extern "C" void kernel_launch(void* const* d_in, const int* in_sizes, int n_in,
                              void* d_out, int out_size, void* d_ws,
                              size_t ws_size, hipStream_t stream) {
  (void)in_sizes; (void)n_in; (void)out_size; (void)ws_size;
  const float* X_Q = (const float*)d_in[0];
  const float* X_K = (const float*)d_in[1];
  const float* X_V = (const float*)d_in[2];
  const float* Wq  = (const float*)d_in[3];
  const float* bq  = (const float*)d_in[4];
  const float* Wk  = (const float*)d_in[5];
  const float* bk  = (const float*)d_in[6];
  const float* Wv  = (const float*)d_in[7];
  const float* bv  = (const float*)d_in[8];
  float* out = (float*)d_out;

  const size_t NX = (size_t)B_ * N_ * DM;
  const size_t NW = (size_t)H_ * DM * DH;
  const size_t NQ = (size_t)B_ * H_ * N_ * DH;

  char* ws = (char*)d_ws;
  _Float16* XQh = (_Float16*)ws; ws += NX * 2;
  _Float16* XKh = (_Float16*)ws; ws += NX * 2;
  _Float16* XVh = (_Float16*)ws; ws += NX * 2;
  _Float16* Wqt = (_Float16*)ws; ws += NW * 2;
  _Float16* Wkt = (_Float16*)ws; ws += NW * 2;
  _Float16* Wvt = (_Float16*)ws; ws += NW * 2;
  _Float16* Qh  = (_Float16*)ws; ws += NQ * 2;
  _Float16* Kh  = (_Float16*)ws; ws += NQ * 2;
  _Float16* Vt  = (_Float16*)ws; ws += NQ * 2;   // [b][h][e][n]

  cvt_f32_f16_kernel<<<2048, 256, 0, stream>>>(X_Q, XQh, (int)NX);
  cvt_f32_f16_kernel<<<2048, 256, 0, stream>>>(X_K, XKh, (int)NX);
  cvt_f32_f16_kernel<<<2048, 256, 0, stream>>>(X_V, XVh, (int)NX);
  wtr_kernel<<<1024, 256, 0, stream>>>(Wq, Wqt);
  wtr_kernel<<<1024, 256, 0, stream>>>(Wk, Wkt);
  wtr_kernel<<<1024, 256, 0, stream>>>(Wv, Wvt);

  proj_kernel<<<1024, 128, 0, stream>>>(XQh, Wqt, bq, Qh, 0, 0.125f);
  proj_kernel<<<1024, 128, 0, stream>>>(XKh, Wkt, bk, Kh, 0, 1.0f);
  proj_kernel<<<1024, 128, 0, stream>>>(XVh, Wvt, bv, Vt, 1, 1.0f);

  attn_kernel<<<2048, 64, 0, stream>>>(Qh, Kh, Vt, out);
}
